// RSNN_53403623358869
// MI455X (gfx1250) — matrix operand order repaired and checked by reading, not compiled, not run on hardware
//
#include <hip/hip_runtime.h>
#include <hip/hip_bf16.h>
#include <math.h>

// ---------------------------------------------------------------------------
// RSNN forward:  out[b][o] = mean_t o_spk
//   1) A[n][h] = f16(x) @ f16(W_ih^T)  -- one big WMMA GEMM (90% of FLOPs),
//      software-pipelined, double-buffered LDS, async B-tile copies.
//   2) 50 sequential steps: hm = A_t (WMMA C-init) + spk @ W_hh^T (WMMA),
//      LIF update; tiny 256->10 readout per step.
// ---------------------------------------------------------------------------

typedef __attribute__((ext_vector_type(16))) _Float16 v16h;
typedef __attribute__((ext_vector_type(8)))  float    v8f;
typedef __attribute__((ext_vector_type(4)))  unsigned int u32x4;
typedef __attribute__((ext_vector_type(4)))  int      i32x4;
typedef __attribute__((ext_vector_type(4)))  float    f32x4;

// address-space-typed pointees for the gfx1250 async global->LDS builtin
typedef __attribute__((address_space(1))) i32x4 gas_i32x4;   // global
typedef __attribute__((address_space(3))) i32x4 las_i32x4;   // LDS

union Frag16 { v16h h; u32x4 u[2]; };
union Pack8  { _Float16 h[8];  u32x4 u; };
union PackB  { _Float16 h[32]; u32x4 u[4]; };

#if defined(__has_builtin)
#  if __has_builtin(__builtin_amdgcn_global_load_async_to_lds_b128) && \
      __has_builtin(__builtin_amdgcn_s_wait_asynccnt)
#    define HAVE_ASYNC_LDS 1
#  endif
#endif
#ifndef HAVE_ASYNC_LDS
#  define HAVE_ASYNC_LDS 0
#endif

#define THRESH 0.3f
#define KC   32      // K chunk
#define KCP  40      // padded LDS row stride (halves): 80B, 16B-aligned, conflict-free
#define BM   64      // workgroup M tile
#define HN   256     // N (= hidden size), handled fully per workgroup

// --------------------------- weight conversion -----------------------------
__global__ __launch_bounds__(256) void cvt_w_f16(const float* __restrict__ src,
                                                 _Float16* __restrict__ dst,
                                                 int rows, int K, int Kp) {
    int idx = blockIdx.x * 256 + threadIdx.x;
    if (idx >= rows * Kp) return;
    int r = idx / Kp, k = idx - r * Kp;
    dst[idx] = (k < K) ? (_Float16)src[(size_t)r * K + k] : (_Float16)0.0f;
}

// ------------------------------ state init ---------------------------------
__global__ __launch_bounds__(256) void init_state(_Float16* hs0, float* hmem,
                                                  float* omem, float* ospk,
                                                  float* osum) {
    int idx = blockIdx.x * 256 + threadIdx.x;     // 65536 threads
    hs0[idx]  = (_Float16)0.0f;
    hmem[idx] = 0.0f;
    if (idx < 2560) { omem[idx] = 0.0f; ospk[idx] = 0.0f; osum[idx] = 0.0f; }
}

// ---------------- Phase A: A = x @ W_ih^T   (M=12800,N=256,K=2312) ---------
__global__ __launch_bounds__(256) void gemm_ih(const float* __restrict__ x,
                                               const _Float16* __restrict__ wih, // [256][Kp] f16
                                               float* __restrict__ A,
                                               int M, int K, int Kp) {
    __shared__ __attribute__((aligned(16))) _Float16 As[2][BM * KCP];
    __shared__ __attribute__((aligned(16))) _Float16 Bs[2][HN * KCP];

    const int tid  = threadIdx.x;
    const int lane = tid & 31;
    const int wid  = tid >> 5;
    const int wm   = wid & 1;        // 2 wave-rows  (32 M each)
    const int wn   = wid >> 1;       // 4 wave-cols  (64 N each)
    const int m0   = blockIdx.x * BM;
    const int lr   = lane & 15;
    const int hi   = lane >> 4;
    const int ra   = tid >> 2;              // A-tile row this thread fills
    const int ka   = (tid & 3) * 8;         // 8 halves per thread
    const int nchunks = Kp / KC;            // 73

    v8f acc[2][4];
    const v8f vz = {0.f,0.f,0.f,0.f,0.f,0.f,0.f,0.f};
#pragma unroll
    for (int i = 0; i < 2; ++i)
#pragma unroll
        for (int j = 0; j < 4; ++j) acc[i][j] = vz;

    // x row f32 -> 8 f16, uniform fast path except last chunk
    auto loadA = [&](int c, Pack8& p) {
        const int k0 = c * KC;
        const float* src = x + (size_t)(m0 + ra) * K + k0 + ka;
        if (k0 + KC <= K) {                       // whole chunk in-bounds
            f32x4 v0 = *(const f32x4*)(src);
            f32x4 v1 = *(const f32x4*)(src + 4);
#pragma unroll
            for (int i = 0; i < 4; ++i) {
                p.h[i]     = (_Float16)v0[i];
                p.h[4 + i] = (_Float16)v1[i];
            }
        } else {                                   // K tail (last chunk only)
#pragma unroll
            for (int i = 0; i < 8; ++i)
                p.h[i] = ((k0 + ka + i) < K) ? (_Float16)src[i] : (_Float16)0.0f;
        }
    };

#if HAVE_ASYNC_LDS
    // weights: 64B/thread contiguous -> async global->LDS (ASYNCcnt path)
    auto issueB = [&](int c, int buf) {
        const _Float16* g = wih + (size_t)tid * Kp + c * KC;
        _Float16* l = &Bs[buf][tid * KCP];
#pragma unroll
        for (int q = 0; q < 4; ++q)
            __builtin_amdgcn_global_load_async_to_lds_b128(
                (gas_i32x4*)(g + q * 8), (las_i32x4*)(l + q * 8), 0, 0);
    };
#else
    auto loadB = [&](int c, PackB& p) {
        const u32x4* src = (const u32x4*)(wih + (size_t)tid * Kp + c * KC);
#pragma unroll
        for (int q = 0; q < 4; ++q) p.u[q] = src[q];
    };
    auto commitB = [&](int buf, const PackB& p) {
        u32x4* dst = (u32x4*)&Bs[buf][tid * KCP];
#pragma unroll
        for (int q = 0; q < 4; ++q) dst[q] = p.u[q];
    };
#endif

    // ---- prologue: fill buffer 0 ----
    {
        Pack8 a0; loadA(0, a0);
        *(u32x4*)&As[0][ra * KCP + ka] = a0.u;
#if HAVE_ASYNC_LDS
        issueB(0, 0);
        __builtin_amdgcn_s_wait_asynccnt(0);
#else
        PackB b0; loadB(0, b0); commitB(0, b0);
#endif
    }
    __syncthreads();

    // ---- main pipeline: one barrier per chunk ----
    for (int c = 0; c < nchunks; ++c) {
        const int  cur  = c & 1;
        const bool more = (c + 1 < nchunks);

        Pack8 aN;
#if !HAVE_ASYNC_LDS
        PackB bN;
#endif
        if (more) {
            loadA(c + 1, aN);                     // overlap with WMMA below
#if HAVE_ASYNC_LDS
            issueB(c + 1, cur ^ 1);               // async into other buffer
#else
            loadB(c + 1, bN);
#endif
        }

        Frag16 a[2], b[4];
#pragma unroll
        for (int i = 0; i < 2; ++i) {             // A frag: lane<16 -> K{0..7,16..23}
            const _Float16* p = &As[cur][(wm * 32 + i * 16 + lr) * KCP + hi * 8];
            a[i].u[0] = *(const u32x4*)(p);
            a[i].u[1] = *(const u32x4*)(p + 16);
        }
#pragma unroll
        for (int j = 0; j < 4; ++j) {             // B frag: lane<16 -> K{0..7,16..23}, n=lane%16
            const _Float16* p = &Bs[cur][(wn * 64 + j * 16 + lr) * KCP + hi * 8];
            b[j].u[0] = *(const u32x4*)(p);
            b[j].u[1] = *(const u32x4*)(p + 16);
        }
#pragma unroll
        for (int i = 0; i < 2; ++i)
#pragma unroll
            for (int j = 0; j < 4; ++j)
                acc[i][j] = __builtin_amdgcn_wmma_f32_16x16x32_f16(
                    false, a[i].h, false, b[j].h, (short)0, acc[i][j], false, false);

        if (more) {
            *(u32x4*)&As[cur ^ 1][ra * KCP + ka] = aN.u;
#if !HAVE_ASYNC_LDS
            commitB(cur ^ 1, bN);
#endif
        }
#if HAVE_ASYNC_LDS
        __builtin_amdgcn_s_wait_asynccnt(0);
#endif
        __syncthreads();
    }

    // store: VGPR v, lane<16 -> M=v, lane>=16 -> M=8+v; N = base + lane%16
#pragma unroll
    for (int i = 0; i < 2; ++i)
#pragma unroll
        for (int j = 0; j < 4; ++j)
#pragma unroll
            for (int v = 0; v < 8; ++v) {
                int m = m0 + wm * 32 + i * 16 + hi * 8 + v;
                int n = wn * 64 + j * 16 + lr;
                A[(size_t)m * HN + n] = acc[i][j][v];
            }
}

// ------------- Phase H (per t): hm = A_t + spk @ W_hh^T, LIF ---------------
__global__ __launch_bounds__(256) void step_hidden(const float* __restrict__ A,
                                                   const _Float16* __restrict__ whh, // [256][256] f16
                                                   const _Float16* __restrict__ hs_old,
                                                   _Float16* __restrict__ hs_new,
                                                   float* __restrict__ hmem,
                                                   int t, int T, float alpha) {
    __shared__ __attribute__((aligned(16))) _Float16 As[BM * KCP];
    __shared__ __attribute__((aligned(16))) _Float16 Bs[HN * KCP];

    const int tid  = threadIdx.x;
    const int lane = tid & 31;
    const int wid  = tid >> 5;
    const int wm   = wid & 1;
    const int wn   = wid >> 1;
    const int b0   = blockIdx.x * BM;     // batch-row base
    const int lr   = lane & 15;
    const int hi   = lane >> 4;

    // C-init: precomputed input current A[(b*T + t)][h]
    v8f acc[2][4];
#pragma unroll
    for (int i = 0; i < 2; ++i)
#pragma unroll
        for (int j = 0; j < 4; ++j)
#pragma unroll
            for (int v = 0; v < 8; ++v) {
                int b = b0 + wm * 32 + i * 16 + hi * 8 + v;
                int n = wn * 64 + j * 16 + lr;
                acc[i][j][v] = A[((size_t)b * T + t) * HN + n];
            }

    const int ra = tid >> 2;
    const int ka = (tid & 3) * 8;

    for (int c = 0; c < HN / KC; ++c) {   // 8 chunks
        const int k0 = c * KC;
        // A tile: spikes, already f16, straight 16B copy
        *(u32x4*)&As[ra * KCP + ka] =
            *(const u32x4*)(hs_old + (size_t)(b0 + ra) * HN + k0 + ka);
        // B tile
        {
            const u32x4* src = (const u32x4*)(whh + (size_t)tid * HN + k0);
            u32x4* dst = (u32x4*)&Bs[tid * KCP];
            dst[0] = src[0]; dst[1] = src[1]; dst[2] = src[2]; dst[3] = src[3];
        }
        __syncthreads();

        Frag16 a[2], b[4];
#pragma unroll
        for (int i = 0; i < 2; ++i) {
            const _Float16* p = &As[(wm * 32 + i * 16 + lr) * KCP + hi * 8];
            a[i].u[0] = *(const u32x4*)(p);
            a[i].u[1] = *(const u32x4*)(p + 16);
        }
#pragma unroll
        for (int j = 0; j < 4; ++j) {
            const _Float16* p = &Bs[(wn * 64 + j * 16 + lr) * KCP + hi * 8];
            b[j].u[0] = *(const u32x4*)(p);
            b[j].u[1] = *(const u32x4*)(p + 16);
        }
#pragma unroll
        for (int i = 0; i < 2; ++i)
#pragma unroll
            for (int j = 0; j < 4; ++j)
                acc[i][j] = __builtin_amdgcn_wmma_f32_16x16x32_f16(
                    false, a[i].h, false, b[j].h, (short)0, acc[i][j], false, false);
        __syncthreads();
    }

    // LIF epilogue: hm = acc + alpha*mem*(1-spk_old); spike/gate; write state
#pragma unroll
    for (int i = 0; i < 2; ++i)
#pragma unroll
        for (int j = 0; j < 4; ++j)
#pragma unroll
            for (int v = 0; v < 8; ++v) {
                int b = b0 + wm * 32 + i * 16 + hi * 8 + v;
                int n = wn * 64 + j * 16 + lr;
                size_t idx = (size_t)b * HN + n;
                float so = (float)hs_old[idx];
                float hm = acc[i][j][v] + alpha * hmem[idx] * (1.0f - so);
                hmem[idx]   = (hm < THRESH) ? hm : 0.0f;
                hs_new[idx] = (_Float16)((hm > THRESH) ? 1.0f : 0.0f);
            }
}

// ----------------- Phase O (per t): readout LIF, O=10 ----------------------
__global__ __launch_bounds__(256) void step_out(const _Float16* __restrict__ hs,
                                                const float* __restrict__ who, // [10][256]
                                                float* __restrict__ omem,
                                                float* __restrict__ ospk,
                                                float* __restrict__ osum,
                                                float* __restrict__ out,
                                                int t, int T, float alpha) {
    __shared__ float wl[10 * 256];
    const int tid = threadIdx.x;               // one thread per batch row
    for (int i = tid; i < 10 * 256; i += 256) wl[i] = who[i];
    __syncthreads();

    float dot[10];
#pragma unroll
    for (int j = 0; j < 10; ++j) dot[j] = 0.0f;
    for (int k = 0; k < 256; ++k) {
        float s = (float)hs[(size_t)tid * 256 + k];
#pragma unroll
        for (int j = 0; j < 10; ++j) dot[j] += s * wl[j * 256 + k];
    }
#pragma unroll
    for (int j = 0; j < 10; ++j) {
        size_t idx = (size_t)tid * 10 + j;
        float om = alpha * omem[idx] * (1.0f - ospk[idx]) + dot[j];
        float sp = (om > THRESH) ? 1.0f : 0.0f;
        omem[idx] = (om < THRESH) ? om : 0.0f;
        ospk[idx] = sp;
        float s = osum[idx] + sp;
        osum[idx] = s;
        if (t == T - 1) out[idx] = s / (float)T;
    }
}

// ---------------------------------------------------------------------------
extern "C" void kernel_launch(void* const* d_in, const int* in_sizes, int n_in,
                              void* d_out, int out_size, void* d_ws, size_t ws_size,
                              hipStream_t stream) {
    const int B = 256, T = 50, I = 2312, H = 256;
    const int Kp = 2336;                       // I padded to multiple of 32
    const float alpha = expf(-1.0f / 0.83f);

    const float* x    = (const float*)d_in[0];
    const float* w_ih = (const float*)d_in[1];
    const float* w_hh = (const float*)d_in[2];
    const float* w_ho = (const float*)d_in[3];
    float* out = (float*)d_out;

    // workspace carve-up (~15 MB total)
    char* ws = (char*)d_ws;
    float*     A     = (float*)ws;                        ws += (size_t)B * T * H * 4;  // 13.1 MB
    _Float16*  wih16 = (_Float16*)ws;                     ws += (size_t)H * Kp * 2;
    _Float16*  whh16 = (_Float16*)ws;                     ws += (size_t)H * H * 2;
    _Float16*  hs0   = (_Float16*)ws;                     ws += (size_t)B * H * 2;
    _Float16*  hs1   = (_Float16*)ws;                     ws += (size_t)B * H * 2;
    float*     hmem  = (float*)ws;                        ws += (size_t)B * H * 4;
    float*     omem  = (float*)ws;                        ws += (size_t)B * 10 * 4;
    float*     ospk  = (float*)ws;                        ws += (size_t)B * 10 * 4;
    float*     osum  = (float*)ws;

    // weights -> f16 (zero-padded K), fresh state
    cvt_w_f16<<<(H * Kp + 255) / 256, 256, 0, stream>>>(w_ih, wih16, H, I, Kp);
    cvt_w_f16<<<(H * H + 255) / 256, 256, 0, stream>>>(w_hh, whh16, H, H, H);
    init_state<<<(B * H) / 256, 256, 0, stream>>>(hs0, hmem, omem, ospk, osum);

    // hoisted input-current GEMM: 90% of all FLOPs, one pass over x
    gemm_ih<<<(B * T) / BM, 256, 0, stream>>>(x, wih16, A, B * T, I, Kp);

    // sequential recurrence, spike buffers ping-pong
    for (int t = 0; t < T; ++t) {
        _Float16* so = (t & 1) ? hs1 : hs0;
        _Float16* sn = (t & 1) ? hs0 : hs1;
        step_hidden<<<B / BM, 256, 0, stream>>>(A, whh16, so, sn, hmem, t, T, alpha);
        step_out<<<1, 256, 0, stream>>>(sn, w_ho, omem, ospk, osum, out, t, T, alpha);
    }
}
